// DBM_58660663329193
// MI455X (gfx1250) — matrix operand order repaired and checked by reading, not compiled, not run on hardware
//
#include <hip/hip_runtime.h>
#include <hip/hip_bf16.h>

typedef unsigned short u16;
typedef unsigned int u32;
typedef unsigned long long u64;
typedef __attribute__((ext_vector_type(4)))  u32      u32x4;
typedef __attribute__((ext_vector_type(4)))  int      i32x4;
typedef __attribute__((ext_vector_type(16))) __bf16   v16bf;
typedef __attribute__((ext_vector_type(8)))  float    v8f;

union BF16x16 { v16bf v; u32x4 q[2]; };

#define BM 128
#define BN 256
#define BK 32

#define AS1 __attribute__((address_space(1)))
#define AS3 __attribute__((address_space(3)))

// ---------- small device helpers ----------
__device__ __forceinline__ float bf2f(u16 u) {
  union { u32 i; float f; } v; v.i = ((u32)u) << 16; return v.f;
}
__device__ __forceinline__ u16 f2bf(float f) {
  union { float f; u32 i; } v; v.f = f;
  u32 u = v.i + 0x7FFFu + ((v.i >> 16) & 1u);
  return (u16)(u >> 16);
}
__device__ __forceinline__ float rnd01(u32 seed, u32 idx) {
  u32 x = idx * 0x9E3779B9u ^ (seed * 0x85EBCA6Bu);
  x ^= x >> 16; x *= 0x7FEB352Du; x ^= x >> 15; x *= 0x846CA68Bu; x ^= x >> 16;
  return (float)(x >> 8) * (1.0f / 16777216.0f);
}
__device__ __forceinline__ float sigmoidf(float x) {
  return 1.0f / (1.0f + __expf(-x));
}

// ---------- CDNA5 async global->LDS copy (ASYNCcnt tracked) ----------
__device__ __forceinline__ void async_b128(const u16* g, u16* l) {
#if __has_builtin(__builtin_amdgcn_global_load_async_to_lds_b128)
  // param 1: global (AS1) int4*, param 2: LDS (AS3) int4*
  __builtin_amdgcn_global_load_async_to_lds_b128(
      (AS1 i32x4*)(u64)g, (AS3 i32x4*)(u32)(u64)l, 0, 0);
#else
  unsigned loff = (unsigned)(u64)l;  // generic shared addr truncates to LDS offset
  asm volatile("global_load_async_to_lds_b128 %0, %1, off"
               :: "v"(loff), "v"((u64)g)
               : "memory");
#endif
}
__device__ __forceinline__ void async_wait0() {
#if __has_builtin(__builtin_amdgcn_s_wait_asynccnt)
  __builtin_amdgcn_s_wait_asynccnt(0);
#else
  asm volatile("s_wait_asynccnt 0x0" ::: "memory");
#endif
}

// ---------- conversion / utility kernels ----------
__global__ void cvt_bf16_kernel(const float* __restrict__ src, u16* __restrict__ dst, size_t n) {
  size_t i = (size_t)blockIdx.x * blockDim.x + threadIdx.x;
  if (i < n) dst[i] = f2bf(src[i]);
}

__global__ void cvt_tr_bf16_kernel(const float* __restrict__ src, u16* __restrict__ dst,
                                   int rows, int cols) {
  // dst[c*rows + r] = bf16(src[r*cols + c])
  size_t i = (size_t)blockIdx.x * blockDim.x + threadIdx.x;
  size_t total = (size_t)rows * cols;
  if (i >= total) return;
  int c = (int)(i % cols);
  int r = (int)(i / cols);
  dst[(size_t)c * rows + r] = f2bf(src[i]);
}

__global__ void extract_sign_f32_kernel(const float* __restrict__ v, u16* __restrict__ out, int Bn) {
  int i = blockIdx.x * blockDim.x + threadIdx.x;
  int total = Bn * 2048;
  if (i >= total) return;
  int b = i >> 11, s = i & 2047;
  out[i] = f2bf(v[(size_t)b * 4096 + 2 * s + 1]);
}

__global__ void extract_sign_bf_kernel(const u16* __restrict__ v, u16* __restrict__ out, int Bn) {
  int i = blockIdx.x * blockDim.x + threadIdx.x;
  int total = Bn * 2048;
  if (i >= total) return;
  int b = i >> 11, s = i & 2047;
  out[i] = v[(size_t)b * 4096 + 2 * s + 1];
}

__global__ void build_vneg_kernel(const float* __restrict__ occ, const float* __restrict__ pv,
                                  u16* __restrict__ vneg, int Bn, u32 seed) {
  int i = blockIdx.x * blockDim.x + threadIdx.x;
  int total = Bn * 2048;
  if (i >= total) return;
  int b = i >> 11, s = i & 2047;
  size_t base = (size_t)b * 4096 + 2 * s;
  vneg[base] = f2bf(occ[i]);
  float p = pv[base + 1];
  vneg[base + 1] = (rnd01(seed, (u32)i) < p) ? (u16)0x3F80 : (u16)0;
}

__global__ void fill_zero_kernel(float* __restrict__ p, size_t n) {
  size_t i = (size_t)blockIdx.x * blockDim.x + threadIdx.x;
  size_t stride = (size_t)gridDim.x * blockDim.x;
  for (; i < n; i += stride) p[i] = 0.0f;
}

// out[j] = scale * sum_b (X[b,j] - Y[b,j])
__global__ void colmean_diff_kernel(const u16* __restrict__ X, const u16* __restrict__ Y,
                                    float* __restrict__ out, int rows, int N, float scale) {
  int j = blockIdx.x * blockDim.x + threadIdx.x;
  if (j >= N) return;
  float s = 0.0f;
  for (int b = 0; b < rows; ++b)
    s += bf2f(X[(size_t)b * N + j]) - bf2f(Y[(size_t)b * N + j]);
  out[j] = s * scale;
}

// -db_v_full: odd entries = mean_b(v_neg_sign - v_pos_sign), even entries = 0
__global__ void dbv_kernel(const u16* __restrict__ vneg, const float* __restrict__ vdata,
                           float* __restrict__ out, int rows) {
  int j = blockIdx.x * blockDim.x + threadIdx.x;
  if (j >= 4096) return;
  if ((j & 1) == 0) { out[j] = 0.0f; return; }
  float s = 0.0f;
  for (int b = 0; b < rows; ++b)
    s += bf2f(vneg[(size_t)b * 4096 + j]) - vdata[(size_t)b * 4096 + j];
  out[j] = s * (1.0f / 1024.0f);
}

__global__ void loss_partial_kernel(const float* __restrict__ vdata, const u16* __restrict__ vneg,
                                    float* __restrict__ part, int total) {
  __shared__ float red[256];
  float s = 0.0f;
  const float eps = 1e-7f;
  for (int i = blockIdx.x * 256 + threadIdx.x; i < total; i += gridDim.x * 256) {
    int b = i >> 11, sx = i & 2047;
    size_t o = (size_t)b * 4096 + 2 * sx + 1;
    float st = vdata[o];
    float sp = bf2f(vneg[o]);
    s += -(st * __logf(sp + eps) + (1.0f - st) * __logf(1.0f - sp + eps));
  }
  red[threadIdx.x] = s;
  __syncthreads();
  for (int o = 128; o > 0; o >>= 1) {
    if ((int)threadIdx.x < o) red[threadIdx.x] += red[threadIdx.x + o];
    __syncthreads();
  }
  if (threadIdx.x == 0) part[blockIdx.x] = red[0];
}

__global__ void loss_final_kernel(const float* __restrict__ part, float* __restrict__ out,
                                  int n, float scale) {
  __shared__ float red[256];
  float s = ((int)threadIdx.x < n) ? part[threadIdx.x] : 0.0f;
  red[threadIdx.x] = s;
  __syncthreads();
  for (int o = 128; o > 0; o >>= 1) {
    if ((int)threadIdx.x < o) red[threadIdx.x] += red[threadIdx.x + o];
    __syncthreads();
  }
  if (threadIdx.x == 0) out[0] = red[0] * scale;
}

// ---------- WMMA GEMM (NT): C[M,N] = sum_k A[m,k] * Bt[n,k] ----------
// A: M x K row-major bf16; Bt: N x K row-major bf16 (i.e. B pre-transposed)
// Staging uses CDNA5 async global->LDS copies (no VGPR round-trip).
// Epilogue: x = acc (+ zin[m,n]) (+ bias[n]); optional sigmoid; write f32 and/or
// Bernoulli-sampled bf16 (1.0/0.0).
template <bool ADD_IN, bool BIAS, bool SIG, bool SAMPLE, bool OUTF, bool OUTB>
__global__ __launch_bounds__(256) void gemm_nt(
    const u16* __restrict__ A, const u16* __restrict__ Bt,
    const float* __restrict__ zin, const float* __restrict__ bias,
    float* __restrict__ outf, u16* __restrict__ outb,
    int M, int N, int K, u32 seed) {
  __shared__ __align__(16) u16 sA[2][BM * BK];
  __shared__ __align__(16) u16 sB[2][BN * BK];
  const int t = threadIdx.x;
  const int lane = t & 31, wid = t >> 5;
  const int wm = wid >> 2, wn = wid & 3;
  const int m0 = blockIdx.y * BM, n0 = blockIdx.x * BN;
  const int nk = K / BK;

  v8f acc[4][4];
  const v8f vzero = {0.f, 0.f, 0.f, 0.f, 0.f, 0.f, 0.f, 0.f};
#pragma unroll
  for (int i = 0; i < 4; ++i)
#pragma unroll
    for (int j = 0; j < 4; ++j) acc[i][j] = vzero;

  auto stage = [&](int kb, int buf) {
    {  // A tile: 128 rows x 32 halfs; thread -> (row, half-row segment), 2x b128
      int row = t >> 1, seg = t & 1;
      const u16* g = A + (size_t)(m0 + row) * K + kb * BK + seg * 16;
      u16* l = &sA[buf][row * BK + seg * 16];
      async_b128(g, l);
      async_b128(g + 8, l + 8);
      if (kb + 2 < nk) __builtin_prefetch((const void*)(g + 2 * BK), 0, 0);
    }
    {  // Bt tile: 256 rows x 32 halfs; thread t -> row t, 4x b128
      const u16* g = Bt + (size_t)(n0 + t) * K + kb * BK;
      u16* l = &sB[buf][t * BK];
      async_b128(g, l);
      async_b128(g + 8, l + 8);
      async_b128(g + 16, l + 16);
      async_b128(g + 24, l + 24);
      if (kb + 2 < nk) __builtin_prefetch((const void*)(g + 2 * BK), 0, 0);
    }
  };

  stage(0, 0);
  async_wait0();
  __syncthreads();
  for (int kb = 0; kb < nk; ++kb) {
    int cur = kb & 1;
    if (kb + 1 < nk) stage(kb + 1, cur ^ 1);

    BF16x16 fa[4], fb[4];
    const int kh = lane >> 4;  // half-selector per ISA A/B layouts
#pragma unroll
    for (int mi = 0; mi < 4; ++mi) {
      const u16* base = &sA[cur][(wm * 64 + mi * 16 + (lane & 15)) * BK];
      fa[mi].q[0] = *(const u32x4*)(base + kh * 8);        // K = kh*8 .. +7
      fa[mi].q[1] = *(const u32x4*)(base + kh * 8 + 16);   // K = 16+kh*8 .. +7
    }
#pragma unroll
    for (int ni = 0; ni < 4; ++ni) {
      const u16* base = &sB[cur][(wn * 64 + ni * 16 + (lane & 15)) * BK];
      fb[ni].q[0] = *(const u32x4*)(base + kh * 8);        // K = kh*8 .. +7
      fb[ni].q[1] = *(const u32x4*)(base + kh * 8 + 16);   // K = 16+kh*8 .. +7
    }
#pragma unroll
    for (int mi = 0; mi < 4; ++mi)
#pragma unroll
      for (int ni = 0; ni < 4; ++ni)
        acc[mi][ni] = __builtin_amdgcn_wmma_f32_16x16x32_bf16(
            false, fa[mi].v, false, fb[ni].v, (short)0, acc[mi][ni], false, false);

    async_wait0();   // next tile's async copies complete before anyone reuses LDS
    __syncthreads();
  }

#pragma unroll
  for (int mi = 0; mi < 4; ++mi) {
#pragma unroll
    for (int ni = 0; ni < 4; ++ni) {
      int col = n0 + wn * 64 + ni * 16 + (lane & 15);
      int rbase = m0 + wm * 64 + mi * 16 + (lane >> 4) * 8;
#pragma unroll
      for (int r = 0; r < 8; ++r) {
        int row = rbase + r;
        size_t o = (size_t)row * N + col;
        float x = acc[mi][ni][r];
        if (ADD_IN) x += zin[o];
        if (BIAS) x += bias[col];
        if (SIG) x = sigmoidf(x);
        if (OUTF) outf[o] = x;
        if (SAMPLE) outb[o] = (rnd01(seed, (u32)o) < x) ? (u16)0x3F80 : (u16)0;
      }
    }
  }
}

// ---------- WMMA GEMM (TN): C[M,N] = sum_k A[k,m] * B[k,n] ----------
// A: K x M row-major bf16; B: K x N row-major bf16.
// out[m*ldc + n*cstride + coff] = (BETA ? old : 0) + alpha * acc
template <bool BETA>
__global__ __launch_bounds__(256) void gemm_tn(
    const u16* __restrict__ A, const u16* __restrict__ B,
    float* __restrict__ out, int M, int N, int K,
    int ldc, int cstride, int coff, float alpha) {
  __shared__ __align__(16) u16 sA[2][BM * BK];
  __shared__ __align__(16) u16 sB[2][BN * BK];
  const int t = threadIdx.x;
  const int lane = t & 31, wid = t >> 5;
  const int wm = wid >> 2, wn = wid & 3;
  const int m0 = blockIdx.y * BM, n0 = blockIdx.x * BN;
  const int nk = K / BK;

  v8f acc[4][4];
  const v8f vzero = {0.f, 0.f, 0.f, 0.f, 0.f, 0.f, 0.f, 0.f};
#pragma unroll
  for (int i = 0; i < 4; ++i)
#pragma unroll
    for (int j = 0; j < 4; ++j) acc[i][j] = vzero;

  auto stage = [&](int kb, int buf) {
    int k = t & 31, seg = t >> 5;
    {  // A tile: transpose K-major global -> [m][k] LDS
      const u16* g = A + (size_t)(kb * BK + k) * M + m0 + seg * 16;
#pragma unroll
      for (int i = 0; i < 16; ++i) sA[buf][(seg * 16 + i) * BK + k] = g[i];
      if (kb + 2 < nk) __builtin_prefetch((const void*)(g + 2 * (size_t)BK * M), 0, 0);
    }
    {  // B tile: transpose K-major global -> [n][k] LDS (256 n per tile)
      for (int gseg = seg; gseg < 16; gseg += 8) {
        const u16* g = B + (size_t)(kb * BK + k) * N + n0 + gseg * 16;
#pragma unroll
        for (int i = 0; i < 16; ++i) sB[buf][(gseg * 16 + i) * BK + k] = g[i];
      }
    }
  };

  stage(0, 0);
  __syncthreads();
  for (int kb = 0; kb < nk; ++kb) {
    int cur = kb & 1;
    if (kb + 1 < nk) stage(kb + 1, cur ^ 1);

    BF16x16 fa[4], fb[4];
    const int kh = lane >> 4;
#pragma unroll
    for (int mi = 0; mi < 4; ++mi) {
      const u16* base = &sA[cur][(wm * 64 + mi * 16 + (lane & 15)) * BK];
      fa[mi].q[0] = *(const u32x4*)(base + kh * 8);
      fa[mi].q[1] = *(const u32x4*)(base + kh * 8 + 16);
    }
#pragma unroll
    for (int ni = 0; ni < 4; ++ni) {
      const u16* base = &sB[cur][(wn * 64 + ni * 16 + (lane & 15)) * BK];
      fb[ni].q[0] = *(const u32x4*)(base + kh * 8);
      fb[ni].q[1] = *(const u32x4*)(base + kh * 8 + 16);
    }
#pragma unroll
    for (int mi = 0; mi < 4; ++mi)
#pragma unroll
      for (int ni = 0; ni < 4; ++ni)
        acc[mi][ni] = __builtin_amdgcn_wmma_f32_16x16x32_bf16(
            false, fa[mi].v, false, fb[ni].v, (short)0, acc[mi][ni], false, false);
    __syncthreads();
  }

#pragma unroll
  for (int mi = 0; mi < 4; ++mi) {
#pragma unroll
    for (int ni = 0; ni < 4; ++ni) {
      int ncol = n0 + wn * 64 + ni * 16 + (lane & 15);
      int rbase = m0 + wm * 64 + mi * 16 + (lane >> 4) * 8;
#pragma unroll
      for (int r = 0; r < 8; ++r) {
        int mrow = rbase + r;
        size_t o = (size_t)mrow * ldc + (size_t)ncol * cstride + coff;
        float v = alpha * acc[mi][ni][r];
        out[o] = BETA ? (out[o] + v) : v;
      }
    }
  }
}

// ---------- host orchestration ----------
extern "C" void kernel_launch(void* const* d_in, const int* in_sizes, int n_in,
                              void* d_out, int out_size, void* d_ws, size_t ws_size,
                              hipStream_t stream) {
  (void)in_sizes; (void)n_in; (void)out_size; (void)ws_size;
  const int Bn = 1024, NV = 4096, NH1 = 4096, NH2 = 4096, NS = 2048;
  const float invB = 1.0f / 1024.0f;

  const float* v_data = (const float*)d_in[0];
  const float* occ    = (const float*)d_in[1];
  const float* W1     = (const float*)d_in[2];
  const float* b_v    = (const float*)d_in[3];
  const float* b_h1   = (const float*)d_in[4];
  const float* W2     = (const float*)d_in[5];
  const float* b_h2   = (const float*)d_in[6];
  float* out = (float*)d_out;

  // output layout (flat f32): loss | dW1(16.7M) | db_v(4096) | db_h1(4096) | dW2(16.7M) | db_h2(4096)
  const size_t o_dW1  = 1;
  const size_t o_dbv  = o_dW1 + (size_t)NH1 * NV;
  const size_t o_dbh1 = o_dbv + NV;
  const size_t o_dW2  = o_dbh1 + NH1;
  const size_t o_dbh2 = o_dW2 + (size_t)NH1 * NH2;

  // workspace carve
  char* w = (char*)d_ws;
  auto carve = [&](size_t bytes) -> void* {
    void* p = (void*)w;
    w += (bytes + 255) & ~(size_t)255;
    return p;
  };
  u16* W1b   = (u16*)carve((size_t)NH1 * NV * 2);
  u16* W1Tb  = (u16*)carve((size_t)NV * NH1 * 2);
  u16* W2b   = (u16*)carve((size_t)NH1 * NH2 * 2);
  u16* W2Tb  = (u16*)carve((size_t)NH2 * NH1 * 2);
  u16* vposb = (u16*)carve((size_t)Bn * NV * 2);
  u16* h1db  = (u16*)carve((size_t)Bn * NH1 * 2);
  u16* h2db  = (u16*)carve((size_t)Bn * NH2 * 2);
  u16* vnegb = (u16*)carve((size_t)Bn * NV * 2);
  u16* h1nb  = (u16*)carve((size_t)Bn * NH1 * 2);
  u16* h2nb  = (u16*)carve((size_t)Bn * NH2 * 2);
  u16* h1fb  = (u16*)carve((size_t)Bn * NH1 * 2);
  u16* vspb  = (u16*)carve((size_t)Bn * NS * 2);
  u16* vsnb  = (u16*)carve((size_t)Bn * NS * 2);
  float* z   = (float*)carve((size_t)Bn * NH1 * 4);
  float* pv  = (float*)carve((size_t)Bn * NV * 4);
  float* part = (float*)carve(256 * 4);

  const size_t nW = (size_t)NH1 * NV;      // 16.7M
  const size_t nAct = (size_t)Bn * NV;     // 4.2M
  const int T = 256;

  // weight / input conversion
  cvt_bf16_kernel<<<dim3((nW + T - 1) / T), T, 0, stream>>>(W1, W1b, nW);
  cvt_tr_bf16_kernel<<<dim3((nW + T - 1) / T), T, 0, stream>>>(W1, W1Tb, NH1, NV);
  cvt_bf16_kernel<<<dim3((nW + T - 1) / T), T, 0, stream>>>(W2, W2b, nW);
  cvt_tr_bf16_kernel<<<dim3((nW + T - 1) / T), T, 0, stream>>>(W2, W2Tb, NH1, NH2);
  cvt_bf16_kernel<<<dim3((nAct + T - 1) / T), T, 0, stream>>>(v_data, vposb, nAct);
  extract_sign_f32_kernel<<<dim3((Bn * NS + T - 1) / T), T, 0, stream>>>(v_data, vspb, Bn);

  dim3 gFwd(NH1 / BN, Bn / BM);  // (16, 8)

  // positive phase: h1_data = bern(sigmoid(v@W1^T + b_h1)); h2_data = bern(sigmoid(h1@W2^T + b_h2))
  gemm_nt<false, true, true, true, false, true><<<gFwd, T, 0, stream>>>(
      vposb, W1b, nullptr, b_h1, nullptr, h1db, Bn, NH1, NV, 101u);
  gemm_nt<false, true, true, true, false, true><<<gFwd, T, 0, stream>>>(
      h1db, W2b, nullptr, b_h2, nullptr, h2db, Bn, NH2, NH1, 202u);

  // negative phase, k = 2 Gibbs sweeps
  const u16* vcur = vposb;
  const u16* h2cur = h2db;
  for (int it = 0; it < 2; ++it) {
    u32 s = 1000u + 10u * (u32)it;
    // z = h2_neg @ W2   (Bt = W2^T row-major)
    gemm_nt<false, false, false, false, true, false><<<gFwd, T, 0, stream>>>(
        h2cur, W2Tb, nullptr, nullptr, z, nullptr, Bn, NH1, NH2, 0u);
    // h1_neg = bern(sigmoid(v_neg@W1^T + b_h1 + z))
    gemm_nt<true, true, true, true, false, true><<<gFwd, T, 0, stream>>>(
        vcur, W1b, z, b_h1, nullptr, h1nb, Bn, NH1, NV, s + 0u);
    // h2_neg = bern(sigmoid(h1_neg@W2^T + b_h2))
    gemm_nt<false, true, true, true, false, true><<<gFwd, T, 0, stream>>>(
        h1nb, W2b, nullptr, b_h2, nullptr, h2nb, Bn, NH2, NH1, s + 1u);
    // pv = sigmoid(h1_neg@W1 + b_v)   (Bt = W1^T row-major)
    gemm_nt<false, true, true, false, true, false><<<gFwd, T, 0, stream>>>(
        h1nb, W1Tb, nullptr, b_v, pv, nullptr, Bn, NV, NH1, 0u);
    // v_neg = interleave(occupant, bern(pv[:,1::2]))
    build_vneg_kernel<<<dim3((Bn * NS + T - 1) / T), T, 0, stream>>>(occ, pv, vnegb, Bn, s + 2u);
    vcur = vnegb;
    h2cur = h2nb;
  }

  // final: h1_neg_final = bern(sigmoid(v_neg@W1^T + b_h1 + h2_neg@W2))
  gemm_nt<false, false, false, false, true, false><<<gFwd, T, 0, stream>>>(
      h2nb, W2Tb, nullptr, nullptr, z, nullptr, Bn, NH1, NH2, 0u);
  gemm_nt<true, true, true, true, false, true><<<gFwd, T, 0, stream>>>(
      vnegb, W1b, z, b_h1, nullptr, h1fb, Bn, NH1, NV, 999u);
  extract_sign_bf_kernel<<<dim3((Bn * NS + T - 1) / T), T, 0, stream>>>(vnegb, vsnb, Bn);

  // -dW1_full: zeros everywhere, odd columns get invB*(neg - pos) outer products
  fill_zero_kernel<<<dim3(2048), T, 0, stream>>>(out + o_dW1, nW);
  dim3 gW1(NS / BN, NH1 / BM);   // (8, 32)
  gemm_tn<true><<<gW1, T, 0, stream>>>(h1db, vspb, out + o_dW1,
                                       NH1, NS, Bn, NV, 2, 1, -invB);
  gemm_tn<true><<<gW1, T, 0, stream>>>(h1fb, vsnb, out + o_dW1,
                                       NH1, NS, Bn, NV, 2, 1, +invB);

  // -dW2 = invB*(h1f^T h2_neg - h1_data^T h2_data)
  dim3 gW2(NH2 / BN, NH1 / BM);  // (16, 32)
  gemm_tn<false><<<gW2, T, 0, stream>>>(h1db, h2db, out + o_dW2,
                                        NH1, NH2, Bn, NH2, 1, 0, -invB);
  gemm_tn<true><<<gW2, T, 0, stream>>>(h1fb, h2nb, out + o_dW2,
                                       NH1, NH2, Bn, NH2, 1, 0, +invB);

  // bias gradients (negated per output spec)
  dbv_kernel<<<dim3(NV / T), T, 0, stream>>>(vnegb, v_data, out + o_dbv, Bn);
  colmean_diff_kernel<<<dim3(NH1 / T), T, 0, stream>>>(h1fb, h1db, out + o_dbh1, Bn, NH1, invB);
  colmean_diff_kernel<<<dim3(NH2 / T), T, 0, stream>>>(h2nb, h2db, out + o_dbh2, Bn, NH2, invB);

  // loss = mean BCE over sign sites
  loss_partial_kernel<<<dim3(256), T, 0, stream>>>(v_data, vnegb, part, Bn * NS);
  loss_final_kernel<<<dim3(1), T, 0, stream>>>(part, out, 256, 1.0f / (float)(Bn * NS));
}
